// HdcMambaBlock_68195490726184
// MI455X (gfx1250) — matrix operand order repaired and checked by reading, not compiled, not run on hardware
//
#include <hip/hip_runtime.h>
#include <hip/hip_bf16.h>
#include <math.h>

#define DM 1024
#define DS 16
#define KW 4
#define GROUPS 8
#define BATCH 4
#define SEQ 2048
#define NROWS (BATCH * SEQ)          /* 8192 */
#define PROJ_DIM (3 * DS + 4 * DM)   /* 4144 */
#define K2 (DS + DM)                 /* 1040 */
#define K2PAD 1056                   /* padded to multiple of 32 */
#define EPS 1e-5f
#define SCHUNK 8                     /* scan flush chunk */

typedef __bf16 bf16_t;
typedef __attribute__((ext_vector_type(16))) __bf16 v16bf;
typedef __attribute__((ext_vector_type(8)))  __bf16 v8bf;
typedef __attribute__((ext_vector_type(8)))  float  v8f;

static __device__ __forceinline__ v16bf make_v16(v8bf lo, v8bf hi) {
  union { v16bf v; v8bf h[2]; } u;
  u.h[0] = lo; u.h[1] = hi;
  return u.v;
}

// gfx1250 async global->LDS copy (ASYNCcnt-tracked, no VGPR staging).
// LDS byte offset = low 32 bits of the generic pointer to a __shared__ object.
static __device__ __forceinline__ void async_ld_b128(void* lds, const void* g) {
  asm volatile("global_load_async_to_lds_b128 %0, %1, off"
               :: "v"((unsigned int)(unsigned long long)lds), "v"(g)
               : "memory");
}
static __device__ __forceinline__ void wait_async0() {
  asm volatile("s_wait_asynccnt 0x0" ::: "memory");
}

// ---------------------------------------------------------------------------
// LayerNorm(u) -> bf16 h.  One block per row of 1024.
// ---------------------------------------------------------------------------
__global__ __launch_bounds__(256) void k_layernorm(
    const float* __restrict__ u, const float* __restrict__ g,
    const float* __restrict__ b, bf16_t* __restrict__ h) {
  int row = blockIdx.x;
  const float* ur = u + (size_t)row * DM;
  float v[4];
  float s = 0.f, s2 = 0.f;
  for (int i = 0; i < 4; ++i) {
    v[i] = ur[threadIdx.x + 256 * i];
    s += v[i];
    s2 += v[i] * v[i];
  }
  for (int m = 16; m >= 1; m >>= 1) {
    s  += __shfl_xor(s,  m);
    s2 += __shfl_xor(s2, m);
  }
  __shared__ float r0[8], r1[8];
  __shared__ float mu_s, rs_s;
  int wave = threadIdx.x >> 5;
  if ((threadIdx.x & 31) == 0) { r0[wave] = s; r1[wave] = s2; }
  __syncthreads();
  if (threadIdx.x == 0) {
    float ts = 0.f, ts2 = 0.f;
    for (int i = 0; i < 8; ++i) { ts += r0[i]; ts2 += r1[i]; }
    float mu  = ts / (float)DM;
    float var = ts2 / (float)DM - mu * mu;
    mu_s = mu;
    rs_s = rsqrtf(var + EPS);
  }
  __syncthreads();
  float mu = mu_s, rs = rs_s;
  for (int i = 0; i < 4; ++i) {
    int d = threadIdx.x + 256 * i;
    h[(size_t)row * DM + d] = (bf16_t)((v[i] - mu) * rs * g[d] + b[d]);
  }
}

// ---------------------------------------------------------------------------
// Weight casts
// ---------------------------------------------------------------------------
__global__ void k_cast_bf16(const float* __restrict__ src,
                            bf16_t* __restrict__ dst, int n) {
  int i = blockIdx.x * 256 + threadIdx.x;
  if (i < n) dst[i] = (bf16_t)src[i];
}

__global__ void k_cast_w2(const float* __restrict__ src,
                          bf16_t* __restrict__ dst) {
  int i = blockIdx.x * 256 + threadIdx.x;  // over DM*K2PAD
  int o = i / K2PAD, k = i % K2PAD;
  dst[i] = (k < K2) ? (bf16_t)src[(size_t)o * K2 + k] : (bf16_t)0.f;
}

// ---------------------------------------------------------------------------
// Tiled bf16 WMMA GEMM:  C[M,N] = A[M,K] * B[N,K]^T (both row-major, bf16),
// f32 accumulate.  Block tile 256x128, 8 waves as 4(M)x2(N), wave tile 64x64
// (16 f32 accumulators -> 1.0 ds_load per WMMA).  Double-buffered LDS filled
// by async global->LDS DMA; one barrier per k-step.
// Optional fused epilogue: C += res[., res_off + n] + uadd[., n].
// ---------------------------------------------------------------------------
__global__ __launch_bounds__(256) void k_gemm_bf16(
    const bf16_t* __restrict__ A, int lda,
    const bf16_t* __restrict__ B, int ldb, int Ntot,
    float* __restrict__ C, int ldc, int Ktot,
    const float* __restrict__ res, int res_ld, int res_off,
    const float* __restrict__ uadd) {
  __shared__ bf16_t As[2][256][40];   // 40 KB: 16B-aligned chunks, pad vs conflicts
  __shared__ bf16_t Bs[2][128][40];   // 20 KB

  int tid  = threadIdx.x;
  int wave = tid >> 5, lane = tid & 31;
  int wm = wave >> 1, wn = wave & 1;          // 4(M) x 2(N) wave grid
  int bM = blockIdx.y * 256, bN = blockIdx.x * 128;
  int half = lane >> 4, lm = lane & 15;

  v8f acc[16] = {};

  int r  = tid >> 1;          // 0..127
  int cb = (tid & 1) * 16;    // 0 or 16 : k-chunk within tile
  bool bn_ok = (bN + r) < Ntot;

  const bf16_t* aRow0 = A + (size_t)(bM + r) * lda + cb;
  const bf16_t* aRow1 = A + (size_t)(bM + 128 + r) * lda + cb;
  const bf16_t* bRow  = B + (size_t)(bN + r) * ldb + cb;

  // ---- prologue: zero OOB B rows (persist across both buffers), stage k=0 ----
  if (!bn_ok) {
    v8bf z = {};
    *(v8bf*)&Bs[0][r][cb]     = z;
    *(v8bf*)&Bs[0][r][cb + 8] = z;
    *(v8bf*)&Bs[1][r][cb]     = z;
    *(v8bf*)&Bs[1][r][cb + 8] = z;
  }
  async_ld_b128(&As[0][r][cb],           aRow0);
  async_ld_b128(&As[0][r][cb + 8],       aRow0 + 8);
  async_ld_b128(&As[0][128 + r][cb],     aRow1);
  async_ld_b128(&As[0][128 + r][cb + 8], aRow1 + 8);
  if (bn_ok) {
    async_ld_b128(&Bs[0][r][cb],     bRow);
    async_ld_b128(&Bs[0][r][cb + 8], bRow + 8);
  }
  wait_async0();
  __syncthreads();

  int cur = 0;
  for (int kk = 0; kk < Ktot; kk += 32) {
    // ---- prefetch next k-tile into the other buffer (overlaps WMMAs) ----
    if (kk + 32 < Ktot) {
      int nxt = cur ^ 1;
      const bf16_t* a0 = aRow0 + kk + 32;
      const bf16_t* a1 = aRow1 + kk + 32;
      async_ld_b128(&As[nxt][r][cb],           a0);
      async_ld_b128(&As[nxt][r][cb + 8],       a0 + 8);
      async_ld_b128(&As[nxt][128 + r][cb],     a1);
      async_ld_b128(&As[nxt][128 + r][cb + 8], a1 + 8);
      if (bn_ok) {
        const bf16_t* bn = bRow + kk + 32;
        async_ld_b128(&Bs[nxt][r][cb],     bn);
        async_ld_b128(&Bs[nxt][r][cb + 8], bn + 8);
      }
    }

    // ---- fragments per documented 16-bit layouts; 16 WMMAs ----
    v16bf bfrag[4];
#pragma unroll
    for (int nt = 0; nt < 4; ++nt) {
      const bf16_t* bp = &Bs[cur][wn * 64 + nt * 16 + lm][8 * half];
      bfrag[nt] = make_v16(*(const v8bf*)bp, *(const v8bf*)(bp + 16));
    }
#pragma unroll
    for (int mt = 0; mt < 4; ++mt) {
      const bf16_t* ap = &As[cur][wm * 64 + mt * 16 + lm][8 * half];
      v16bf afrag = make_v16(*(const v8bf*)ap, *(const v8bf*)(ap + 16));
#pragma unroll
      for (int nt = 0; nt < 4; ++nt) {
        acc[mt * 4 + nt] = __builtin_amdgcn_wmma_f32_16x16x32_bf16(
            false, afrag, false, bfrag[nt], (short)0, acc[mt * 4 + nt],
            false, false);
      }
    }

    wait_async0();      // next buffer landed (this wave)
    __syncthreads();    // all waves: reads of cur done, next buffer visible
    cur ^= 1;
  }

  // -- store D (f32 layout: lane n = lane%16, vgpr r -> m = r + 8*(lane/16)) --
#pragma unroll
  for (int mt = 0; mt < 4; ++mt) {
#pragma unroll
    for (int nt = 0; nt < 4; ++nt) {
      int n = bN + wn * 64 + nt * 16 + lm;
      if (n >= Ntot) continue;
      v8f a = acc[mt * 4 + nt];
#pragma unroll
      for (int rr = 0; rr < 8; ++rr) {
        int m = bM + wm * 64 + mt * 16 + half * 8 + rr;
        float val = a[rr];
        if (res) val += res[(size_t)m * res_ld + res_off + n] +
                        uadd[(size_t)m * DM + n];
        C[(size_t)m * ldc + n] = val;
      }
    }
  }
}

// ---------------------------------------------------------------------------
// Depthwise causal conv (KW=4) on x = proj[:,0:DM], + bias, SiLU.
// ---------------------------------------------------------------------------
__global__ void k_conv_silu(const float* __restrict__ proj,
                            const float* __restrict__ cw,
                            const float* __restrict__ cb,
                            float* __restrict__ xc) {
  int i = blockIdx.x * 256 + threadIdx.x;   // over NROWS*DM
  int d = i % DM;
  int bl = i / DM;
  int l = bl % SEQ;
  float acc = cb[d];
#pragma unroll
  for (int j = 0; j < KW; ++j) {
    int ls = l - (KW - 1) + j;
    if (ls >= 0)
      acc += cw[d * KW + j] * proj[(size_t)(bl - l + ls) * PROJ_DIM + d];
  }
  xc[i] = acc / (1.f + __expf(-acc));
}

// ---------------------------------------------------------------------------
// dt = softplus(proj[:, 2*DM : 3*DM])
// ---------------------------------------------------------------------------
__global__ void k_softplus_dt(const float* __restrict__ proj,
                              float* __restrict__ dt) {
  int i = blockIdx.x * 256 + threadIdx.x;
  int d = i % DM;
  size_t row = (size_t)(i / DM);
  float v = proj[row * PROJ_DIM + 2 * DM + d];
  dt[i] = (v > 20.f) ? v : log1pf(__expf(v));
}

// ---------------------------------------------------------------------------
// Selective scan.  Block = (batch b, 32-channel slab), 512 threads.
// Thread (d_local, s): lane = d_local*16 + s.  State h in a register.
// y[b,l,d]     = sum_s h*C + xc*D     (shfl-xor reduce over 16 states)
// s_seq[b,l,s] = (1/DM) sum_d h       (shfl-16 + LDS atomics, flushed per
//                                      SCHUNK steps to cut barrier count 8x)
// ---------------------------------------------------------------------------
__global__ __launch_bounds__(512) void k_scan(
    const float* __restrict__ proj, const float* __restrict__ dt,
    const float* __restrict__ xc, const float* __restrict__ A_log,
    const float* __restrict__ Dp, float* __restrict__ y,
    float* __restrict__ s_seq) {
  int b = blockIdx.y;
  int dblk = blockIdx.x;            // 0..31
  int tid = threadIdx.x;
  int dl = tid >> 4;                // 0..31 channel within slab
  int s  = tid & 15;                // state index
  int d  = dblk * 32 + dl;
  int lane = tid & 31;

  float a  = -__expf(A_log[d * DS + s]);
  float Dv = Dp[d];

  __shared__ float lds_s[SCHUNK][DS];
  if (tid < SCHUNK * DS) lds_s[tid >> 4][tid & 15] = 0.f;
  __syncthreads();

  float h = 0.f;
  size_t rowbase = (size_t)b * SEQ;
  for (int l0 = 0; l0 < SEQ; l0 += SCHUNK) {
#pragma unroll
    for (int lc = 0; lc < SCHUNK; ++lc) {
      size_t row = rowbase + l0 + lc;
      const float* pr = proj + row * PROJ_DIM;
      float dtv = dt[row * DM + d];
      float Bt  = pr[4 * DM + s];
      float Ct  = pr[4 * DM + DS + s];
      float xv  = xc[row * DM + d];

      h = __expf(dtv * a) * h + dtv * Bt * xv;

      float yv = h * Ct;
      yv += __shfl_xor(yv, 1);
      yv += __shfl_xor(yv, 2);
      yv += __shfl_xor(yv, 4);
      yv += __shfl_xor(yv, 8);
      if (s == 0) y[row * DM + d] = yv + xv * Dv;

      float sp = h + __shfl_xor(h, 16);  // this wave's two channels
      if (lane < 16) atomicAdd(&lds_s[lc][s], sp);
    }
    __syncthreads();
    if (tid < SCHUNK * DS) {
      int lc = tid >> 4, ss = tid & 15;
      atomicAdd(&s_seq[(rowbase + l0 + lc) * DS + ss],
                lds_s[lc][ss] * (1.f / (float)DM));
      lds_s[lc][ss] = 0.f;
    }
    __syncthreads();
  }
}

// ---------------------------------------------------------------------------
// GroupNorm stats: one block per (b, group); reduce 128 channels x 2048 steps.
// ---------------------------------------------------------------------------
__global__ __launch_bounds__(256) void k_gn_stats(const float* __restrict__ y,
                                                  float* __restrict__ stats) {
  int bg = blockIdx.x;
  int b = bg >> 3, g = bg & 7;
  float s = 0.f, s2 = 0.f;
  for (int idx = threadIdx.x; idx < 128 * SEQ; idx += 256) {
    int d = g * 128 + (idx & 127);
    int l = idx >> 7;
    float v = y[((size_t)b * SEQ + l) * DM + d];
    s += v;
    s2 += v * v;
  }
  for (int m = 16; m >= 1; m >>= 1) {
    s  += __shfl_xor(s,  m);
    s2 += __shfl_xor(s2, m);
  }
  __shared__ float r0[8], r1[8];
  int wave = threadIdx.x >> 5;
  if ((threadIdx.x & 31) == 0) { r0[wave] = s; r1[wave] = s2; }
  __syncthreads();
  if (threadIdx.x == 0) {
    float ts = 0.f, ts2 = 0.f;
    for (int i = 0; i < 8; ++i) { ts += r0[i]; ts2 += r1[i]; }
    float n   = 128.f * (float)SEQ;
    float mu  = ts / n;
    float var = ts2 / n - mu * mu;
    stats[bg * 2]     = mu;
    stats[bg * 2 + 1] = rsqrtf(var + EPS);
  }
}

// ---------------------------------------------------------------------------
// GroupNorm apply + gate with silu(z), sigmoid(sg)*s_seq, concat + K-pad -> bf16
// ---------------------------------------------------------------------------
__global__ void k_gate_cat(const float* __restrict__ y,
                           const float* __restrict__ proj,
                           const float* __restrict__ s_seq,
                           const float* __restrict__ stats,
                           const float* __restrict__ gg,
                           const float* __restrict__ gb,
                           bf16_t* __restrict__ ycat) {
  int i = blockIdx.x * 256 + threadIdx.x;   // over NROWS*K2PAD
  int f = i % K2PAD;
  size_t row = (size_t)(i / K2PAD);
  int b = (int)(row / SEQ);
  float out;
  if (f < DM) {
    int g = f >> 7;
    float mu = stats[(b * 8 + g) * 2];
    float rs = stats[(b * 8 + g) * 2 + 1];
    float yn = (y[row * DM + f] - mu) * rs * gg[f] + gb[f];
    float z  = proj[row * PROJ_DIM + DM + f];
    out = yn * (z / (1.f + __expf(-z)));
  } else if (f < DM + DS) {
    int sidx = f - DM;
    float sg = proj[row * PROJ_DIM + 4 * DM + 2 * DS + sidx];
    out = s_seq[row * DS + sidx] * (1.f / (1.f + __expf(-sg)));
  } else {
    out = 0.f;  // K padding
  }
  ycat[i] = (bf16_t)out;
}

// ---------------------------------------------------------------------------
extern "C" void kernel_launch(void* const* d_in, const int* in_sizes, int n_in,
                              void* d_out, int out_size, void* d_ws,
                              size_t ws_size, hipStream_t stream) {
  const float* u        = (const float*)d_in[0];
  const float* in_proj  = (const float*)d_in[1];
  const float* conv_w   = (const float*)d_in[2];
  const float* conv_b   = (const float*)d_in[3];
  const float* out_proj = (const float*)d_in[4];
  const float* ln_g     = (const float*)d_in[5];
  const float* ln_b     = (const float*)d_in[6];
  const float* gn_g     = (const float*)d_in[7];
  const float* gn_b     = (const float*)d_in[8];
  const float* A_log    = (const float*)d_in[9];
  const float* Dp       = (const float*)d_in[10];
  float* out = (float*)d_out;
  (void)in_sizes; (void)n_in; (void)out_size; (void)ws_size;

  char* p = (char*)d_ws;
  auto carve = [&](size_t bytes) -> char* {
    char* q = p;
    p += (bytes + 255) & ~(size_t)255;
    return q;
  };
  bf16_t* h_bf  = (bf16_t*)carve((size_t)NROWS * DM * 2);
  bf16_t* w1_bf = (bf16_t*)carve((size_t)PROJ_DIM * DM * 2);
  bf16_t* w2_bf = (bf16_t*)carve((size_t)DM * K2PAD * 2);
  float*  proj  = (float*)carve((size_t)NROWS * PROJ_DIM * 4);
  float*  xc    = (float*)carve((size_t)NROWS * DM * 4);
  float*  dt    = (float*)carve((size_t)NROWS * DM * 4);
  float*  ybuf  = (float*)carve((size_t)NROWS * DM * 4);
  float*  s_seq = (float*)carve((size_t)NROWS * DS * 4);
  bf16_t* ycat  = (bf16_t*)carve((size_t)NROWS * K2PAD * 2);
  float*  stats = (float*)carve((size_t)BATCH * GROUPS * 2 * 4);

  // 1. LayerNorm -> bf16 activations; bf16 weight casts
  k_layernorm<<<NROWS, 256, 0, stream>>>(u, ln_g, ln_b, h_bf);
  k_cast_bf16<<<(PROJ_DIM * DM + 255) / 256, 256, 0, stream>>>(
      in_proj, w1_bf, PROJ_DIM * DM);
  k_cast_w2<<<(DM * K2PAD + 255) / 256, 256, 0, stream>>>(out_proj, w2_bf);

  // 2. in_proj GEMM (8192 x 4144 x 1024), bf16 WMMA + async-LDS double buffer
  k_gemm_bf16<<<dim3(33, 32), 256, 0, stream>>>(
      h_bf, DM, w1_bf, DM, PROJ_DIM, proj, PROJ_DIM, DM,
      nullptr, 0, 0, nullptr);

  // 3. conv+silu, softplus(dt)
  k_conv_silu<<<(NROWS * DM) / 256, 256, 0, stream>>>(proj, conv_w, conv_b, xc);
  k_softplus_dt<<<(NROWS * DM) / 256, 256, 0, stream>>>(proj, dt);

  // 4. selective scan (s_seq accumulated via atomics -> zero first)
  hipMemsetAsync(s_seq, 0, (size_t)NROWS * DS * 4, stream);
  k_scan<<<dim3(32, BATCH), 512, 0, stream>>>(proj, dt, xc, A_log, Dp, ybuf,
                                              s_seq);

  // 5. group norm + gating + concat -> bf16
  k_gn_stats<<<BATCH * GROUPS, 256, 0, stream>>>(ybuf, stats);
  k_gate_cat<<<(NROWS * K2PAD) / 256, 256, 0, stream>>>(
      ybuf, proj, s_seq, stats, gn_g, gn_b, ycat);

  // 6. out_proj GEMM (8192 x 1024 x 1056) with fused +res +u epilogue
  k_gemm_bf16<<<dim3(8, 32), 256, 0, stream>>>(
      ycat, K2PAD, w2_bf, K2PAD, DM, out, DM, K2PAD,
      proj, PROJ_DIM, 3 * DM, u);
}
